// AdvancedMambaAMT_83408264888585
// MI455X (gfx1250) — matrix operand order repaired and checked by reading, not compiled, not run on hardware
//
#include <hip/hip_runtime.h>
#include <hip/hip_bf16.h>
#include <math.h>

// ---------------------------------------------------------------------------
// CDNA5 (gfx1250, wave32) WMMA types
// ---------------------------------------------------------------------------
typedef __bf16 bf16;
typedef __attribute__((ext_vector_type(16))) __bf16 v16bf;
typedef __attribute__((ext_vector_type(8)))  __bf16 v8bf;
typedef __attribute__((ext_vector_type(8)))  float  v8f;

#define WMMA_BF16(a, b, c) \
  __builtin_amdgcn_wmma_f32_16x16x32_bf16(false, (a), false, (b), (short)0, (c), false, false)

__device__ __forceinline__ float gelu_f(float x) {
  return 0.5f * x * (1.0f + erff(x * 0.7071067811865475f));
}
__device__ __forceinline__ float sigmoid_f(float x) {
  return 1.0f / (1.0f + expf(-x));
}
__device__ __forceinline__ v8f zero8() {
  v8f z;
#pragma unroll
  for (int i = 0; i < 8; ++i) z[i] = 0.f;
  return z;
}
// Concatenate two 16-byte bf16 vectors into one WMMA fragment register set.
__device__ __forceinline__ v16bf cat8(v8bf lo, v8bf hi) {
  return __builtin_shufflevector(lo, hi, 0, 1, 2, 3, 4, 5, 6, 7,
                                         8, 9, 10, 11, 12, 13, 14, 15);
}

// ---------------------------------------------------------------------------
// bf16 WMMA GEMM: C[M,N] = epilogue(A[M,K] @ B[K,N]), with B pre-transposed
// to Bt[Npad,K] bf16 so every fragment is two contiguous b128 loads.
//   A fragment (16x32): lane row = lane&15; 16 contiguous bf16 at
//                       k0 + (lane>>4)*16.
//   B fragment (32x16): lane col = lane&15; 16 contiguous bf16 at
//                       Bt[col, k0 + (lane>>4)*16].
//   C/D (f32 16x16):    col = lane&15; row = r + 8*(lane>>4).
// One wave computes a 16x64 strip (4 tiles, A reused 4x); 8 waves per block
// cover 128 rows x 64 cols. K-loop is double-buffered (loads for k+1 issued
// before WMMAs of k) so VMEM overlaps the XDL pipe.
// ---------------------------------------------------------------------------
enum { EP_NONE = 0, EP_BIAS_GELU = 1, EP_BIAS_RES = 2, EP_BIAS = 3,
       EP_GATE = 4, EP_BIAS_SIG = 5, EP_STORE_VT = 6 };

template <int K, int EP, typename OutT>
__global__ __launch_bounds__(256) void gemm_bf16t(
    const bf16* __restrict__ A, const bf16* __restrict__ Bt,
    const float* __restrict__ bias, const float* __restrict__ Rf,
    const bf16* __restrict__ Rb, OutT* __restrict__ C, int M, int N)
{
  const int lane = threadIdx.x & 31;
  const int wave = threadIdx.x >> 5;
  const int half = lane >> 4;
  const int lan  = lane & 15;
  const int row0 = blockIdx.y * 128 + wave * 16;
  const int col0 = blockIdx.x * 64;

  const bf16* ap = A + (size_t)(row0 + lan) * K;
  const bf16* bp0 = Bt + (size_t)(col0 + 0 * 16 + lan) * K + half * 16;
  const bf16* bp1 = Bt + (size_t)(col0 + 1 * 16 + lan) * K + half * 16;
  const bf16* bp2 = Bt + (size_t)(col0 + 2 * 16 + lan) * K + half * 16;
  const bf16* bp3 = Bt + (size_t)(col0 + 3 * 16 + lan) * K + half * 16;

  auto load_a = [&](int k0) -> v16bf {
    return cat8(*(const v8bf*)(ap + k0 + half * 16),
                *(const v8bf*)(ap + k0 + half * 16 + 8));
  };
  auto load_b = [&](const bf16* bp, int k0) -> v16bf {
    const v8bf* p = (const v8bf*)(bp + k0);
    return cat8(p[0], p[1]);
  };

  v8f acc0 = zero8(), acc1 = zero8(), acc2 = zero8(), acc3 = zero8();

  v16bf a  = load_a(0);
  v16bf b0 = load_b(bp0, 0), b1 = load_b(bp1, 0);
  v16bf b2 = load_b(bp2, 0), b3 = load_b(bp3, 0);

#pragma unroll 4
  for (int k0 = 0; k0 < K; k0 += 32) {
    v16bf an = a, bn0 = b0, bn1 = b1, bn2 = b2, bn3 = b3;
    const int kn = k0 + 32;
    if (kn < K) {                       // uniform; resolved at unroll time
      an  = load_a(kn);
      bn0 = load_b(bp0, kn); bn1 = load_b(bp1, kn);
      bn2 = load_b(bp2, kn); bn3 = load_b(bp3, kn);
      if (kn + 32 < K) __builtin_prefetch(ap + kn + 32, 0, 0);
    }
    acc0 = WMMA_BF16(a, b0, acc0);
    acc1 = WMMA_BF16(a, b1, acc1);
    acc2 = WMMA_BF16(a, b2, acc2);
    acc3 = WMMA_BF16(a, b3, acc3);
    a = an; b0 = bn0; b1 = bn1; b2 = bn2; b3 = bn3;
  }

  // ---- epilogue ----
  v8f* accs[4] = { &acc0, &acc1, &acc2, &acc3 };
#pragma unroll
  for (int t = 0; t < 4; ++t) {
    const int col = col0 + t * 16 + lan;
    if (col >= N) continue;             // post-WMMA divergence only
    const float bv = bias ? bias[col] : 0.f;
#pragma unroll
    for (int r = 0; r < 8; ++r) {
      const int    orow = row0 + r + 8 * half;
      const size_t idx  = (size_t)orow * N + col;
      float v = (*accs[t])[r];
      if constexpr (EP == EP_BIAS_GELU)      v = gelu_f(v + bv);
      else if constexpr (EP == EP_BIAS_RES)  v = v + bv + Rf[idx];
      else if constexpr (EP == EP_BIAS)      v = v + bv;
      else if constexpr (EP == EP_GATE)      v = (float)Rb[idx] * sigmoid_f(v + bv);
      else if constexpr (EP == EP_BIAS_SIG)  v = sigmoid_f(v + bv);
      if constexpr (EP == EP_STORE_VT) {
        // scatter into Vt[b][h][d][n] for contiguous P@V fragments
        const size_t b  = (size_t)(orow >> 10);
        const int    n  = orow & 1023;
        const int    h  = col >> 7;
        const int    hd = col & 127;
        C[((b * 4 + h) * 128 + hd) * 1024 + n] = (OutT)v;
      } else {
        C[idx] = (OutT)v;
      }
    }
  }
}

// ---------------------------------------------------------------------------
// Row LayerNorm: reads fp32 rows, writes bf16 rows (optional GELU after).
// ---------------------------------------------------------------------------
template <bool GELU_AFTER>
__global__ __launch_bounds__(256) void ln_rows(
    const float* __restrict__ X, bf16* __restrict__ Y,
    const float* __restrict__ g, const float* __restrict__ bt, int rowlen)
{
  __shared__ float s_sum[256], s_sq[256];
  const float* xr = X + (size_t)blockIdx.x * rowlen;
  bf16*        yr = Y + (size_t)blockIdx.x * rowlen;
  float s = 0.f, q = 0.f;
  for (int i = threadIdx.x; i < rowlen; i += 256) {
    const float v = xr[i];
    s += v; q += v * v;
  }
  s_sum[threadIdx.x] = s;
  s_sq[threadIdx.x]  = q;
  __syncthreads();
  for (int off = 128; off > 0; off >>= 1) {
    if (threadIdx.x < off) {
      s_sum[threadIdx.x] += s_sum[threadIdx.x + off];
      s_sq[threadIdx.x]  += s_sq[threadIdx.x + off];
    }
    __syncthreads();
  }
  const float mean = s_sum[0] / (float)rowlen;
  const float var  = s_sq[0] / (float)rowlen - mean * mean;
  const float rstd = rsqrtf(var + 1e-5f);
  for (int i = threadIdx.x; i < rowlen; i += 256) {
    float v = (xr[i] - mean) * rstd * g[i] + bt[i];
    if constexpr (GELU_AFTER) v = gelu_f(v);
    yr[i] = (bf16)v;
  }
}

// ---------------------------------------------------------------------------
// Weight transpose+convert: Wt[Npad,K] bf16 <- W[K,N] fp32 (zero padded).
// ---------------------------------------------------------------------------
__global__ __launch_bounds__(256) void wt_cvt(
    const float* __restrict__ W, bf16* __restrict__ Wt, int K, int N, int Npad)
{
  const long idx = (long)blockIdx.x * 256 + threadIdx.x;
  if (idx >= (long)Npad * K) return;
  const int n = (int)(idx / K);
  const int k = (int)(idx % K);
  Wt[idx] = (n < N) ? (bf16)W[(long)k * N + n] : (bf16)0.f;
}

__global__ __launch_bounds__(256) void f2bf(
    const float* __restrict__ X, bf16* __restrict__ Y, long n)
{
  const long i = (long)blockIdx.x * 256 + threadIdx.x;
  if (i < n) Y[i] = (bf16)X[i];
}

// ---------------------------------------------------------------------------
// Flash attention, causal + fp32-underflow decay mask:
//   valid(i,j) = (j <= i) && (i + j <= 1032)     [exp(-(i+j)/10) -> 0 in f32]
// Q,K bf16 [B*N,512] row-major (head at col h*128); V bf16 transposed
// Vt[b][h][d][n]. Grid (B*H, N/64); 4 waves/block, wave w owns 16 q-rows.
// ks-loop and V-tile loop are double-buffered; P relayout goes through LDS
// with explicit s_wait_dscnt.
// ---------------------------------------------------------------------------
__global__ __launch_bounds__(128) void attn_flash_wmma(
    const bf16* __restrict__ Q, const bf16* __restrict__ Km,
    const bf16* __restrict__ Vt, bf16* __restrict__ O)
{
  const int lane = threadIdx.x & 31;
  const int wave = threadIdx.x >> 5;
  const int half = lane >> 4;
  const int lan  = lane & 15;
  const int bh = blockIdx.x;
  const int b  = bh >> 2, h = bh & 3;
  const int i0 = blockIdx.y * 64 + wave * 16;
  const long baseQK = ((long)b * 1024) * 512 + h * 128;
  const long baseV  = ((long)(b * 4 + h)) * 128 * 1024;

  // Preload the 4 K-step A-fragments of Q for this 16-row tile.
  v16bf qf[4];
  {
    const bf16* qr = Q + baseQK + (long)(i0 + lan) * 512;
#pragma unroll
    for (int ks = 0; ks < 4; ++ks)
      qf[ks] = cat8(*(const v8bf*)(qr + ks * 32 + half * 16),
                    *(const v8bf*)(qr + ks * 32 + half * 16 + 8));
  }

  v8f accO[8];
#pragma unroll
  for (int t = 0; t < 8; ++t) accO[t] = zero8();
  float mrow[8], lrow[8];
#pragma unroll
  for (int r = 0; r < 8; ++r) { mrow[r] = -__builtin_inff(); lrow[r] = 0.f; }

  __shared__ bf16 pstage[4][16 * 32];
  bf16* ps = pstage[wave];

  const float scale = 0.08838834764831845f;  // 1/sqrt(128)
  const int jmax = i0 + 15;

  for (int j0 = 0; j0 <= jmax; j0 += 32) {
    const bf16* k0p = Km + baseQK + (long)(j0 + lan) * 512;
    const bf16* k1p = Km + baseQK + (long)(j0 + 16 + lan) * 512;
    auto load_k = [&](const bf16* kp, int ks) -> v16bf {
      const v8bf* p = (const v8bf*)(kp + ks * 32 + half * 16);
      return cat8(p[0], p[1]);
    };

    // ---- S = Q K^T, double-buffered over the 4 k-steps -------------------
    v8f s0 = zero8(), s1 = zero8();
    v16bf kf0 = load_k(k0p, 0), kf1 = load_k(k1p, 0);
#pragma unroll
    for (int ks = 0; ks < 4; ++ks) {
      v16bf n0 = kf0, n1 = kf1;
      if (ks < 3) { n0 = load_k(k0p, ks + 1); n1 = load_k(k1p, ks + 1); }
      s0 = WMMA_BF16(qf[ks], kf0, s0);
      s1 = WMMA_BF16(qf[ks], kf1, s1);
      kf0 = n0; kf1 = n1;
    }

    // ---- mask + online softmax ------------------------------------------
#pragma unroll
    for (int r = 0; r < 8; ++r) {
      const int i  = i0 + r + 8 * half;
      const int ja = j0 + lan;
      const int jb = j0 + 16 + lan;
      float v0 = (ja > i || (i + ja) > 1032) ? -__builtin_inff() : s0[r] * scale;
      float v1 = (jb > i || (i + jb) > 1032) ? -__builtin_inff() : s1[r] * scale;
      float m = fmaxf(v0, v1);
#pragma unroll
      for (int off = 1; off < 16; off <<= 1)
        m = fmaxf(m, __shfl_xor(m, off, 32));     // 16 lanes share a row
      const float mnew = fmaxf(mrow[r], m);
      const float corr = expf(mrow[r] - mnew);
      const float p0 = expf(v0 - mnew);
      const float p1 = expf(v1 - mnew);
      float rs = p0 + p1;
#pragma unroll
      for (int off = 1; off < 16; off <<= 1)
        rs += __shfl_xor(rs, off, 32);
      lrow[r] = lrow[r] * corr + rs;
      mrow[r] = mnew;
#pragma unroll
      for (int t = 0; t < 8; ++t) accO[t][r] *= corr;
      s0[r] = p0;
      s1[r] = p1;
    }

    // ---- P: C-layout f32 -> A-layout bf16 via LDS ------------------------
#pragma unroll
    for (int r = 0; r < 8; ++r) {
      ps[(r + 8 * half) * 32 + lan]      = (bf16)s0[r];
      ps[(r + 8 * half) * 32 + 16 + lan] = (bf16)s1[r];
    }
    asm volatile("s_wait_dscnt 0x0" ::: "memory");
    const v16bf pf = cat8(*(const v8bf*)(ps + lan * 32 + half * 16),
                          *(const v8bf*)(ps + lan * 32 + half * 16 + 8));
    asm volatile("s_wait_dscnt 0x0" ::: "memory");

    // ---- O += P @ V, double-buffered over the 8 d-tiles ------------------
    auto load_v = [&](int t) -> v16bf {
      const v8bf* p = (const v8bf*)(Vt + baseV + (long)(t * 16 + lan) * 1024 +
                                    j0 + half * 16);
      return cat8(p[0], p[1]);
    };
    v16bf vf = load_v(0);
#pragma unroll
    for (int t = 0; t < 8; ++t) {
      v16bf vn = vf;
      if (t < 7) vn = load_v(t + 1);
      accO[t] = WMMA_BF16(pf, vf, accO[t]);
      vf = vn;
    }
  }

  // ---- normalize and write bf16 [B*N, 512] ------------------------------
#pragma unroll
  for (int r = 0; r < 8; ++r) {
    const float inv  = 1.0f / lrow[r];
    const long  orow = (long)(b * 1024 + i0 + r + 8 * half);
#pragma unroll
    for (int t = 0; t < 8; ++t)
      O[orow * 512 + h * 128 + t * 16 + lan] = (bf16)(accO[t][r] * inv);
  }
}

// ---------------------------------------------------------------------------
// Host orchestration
// ---------------------------------------------------------------------------
extern "C" void kernel_launch(void* const* d_in, const int* in_sizes, int n_in,
                              void* d_out, int out_size, void* d_ws, size_t ws_size,
                              hipStream_t stream)
{
  (void)in_sizes; (void)n_in; (void)out_size; (void)ws_size;

  const float* x      = (const float*)d_in[0];
  const float* fe_w1  = (const float*)d_in[1];
  const float* fe_b1  = (const float*)d_in[2];
  const float* fe_w2  = (const float*)d_in[3];
  const float* fe_b2  = (const float*)d_in[4];
  const float* fe_g   = (const float*)d_in[5];
  const float* fe_be  = (const float*)d_in[6];
  const float* wq     = (const float*)d_in[7];
  const float* wk     = (const float*)d_in[8];
  const float* wv     = (const float*)d_in[9];
  const float* gate_w = (const float*)d_in[10];
  const float* gate_b = (const float*)d_in[11];
  const float* fc1_w  = (const float*)d_in[12];
  const float* fc1_b  = (const float*)d_in[13];
  const float* ln_g   = (const float*)d_in[14];
  const float* ln_b   = (const float*)d_in[15];
  const float* fc2_w  = (const float*)d_in[16];
  const float* fc2_b  = (const float*)d_in[17];
  float* out = (float*)d_out;

  const int    M  = 16 * 1024;  // B * N rows
  const size_t MB = 1u << 20;
  char* ws = (char*)d_ws;

  // Activations (with region reuse; all reuses are strictly after last read):
  bf16*  xb    = (bf16*)(ws + 0 * MB);     // M x 512  bf16 (x)        [0,16)
  bf16*  h1b   = (bf16*)(ws + 16 * MB);    // M x 1024 bf16            [16,48)
  float* xln   = (float*)(ws + 48 * MB);   // M x 512  f32             [48,80)
  bf16*  xlnb  = (bf16*)(ws + 80 * MB);    // M x 512  bf16            [80,96)
  bf16*  qb    = (bf16*)(ws + 96 * MB);    // M x 512  bf16            [96,112)
  bf16*  kb    = (bf16*)(ws + 112 * MB);   // M x 512  bf16            [112,128)
  bf16*  vtb   = (bf16*)(ws + 128 * MB);   // V transposed bf16        [128,144)
  bf16*  attnb = (bf16*)(ws + 0 * MB);     // reuse xb   (dead after G1)
  bf16*  gatedb= (bf16*)(ws + 16 * MB);    // reuse h1b lower half (dead after G2)
  float* f1    = (float*)(ws + 48 * MB);   // reuse xln  (dead after LN1)
  bf16*  f1b   = (bf16*)(ws + 32 * MB);    // reuse h1b upper half
  // Transposed bf16 weights:
  bf16*  w1t   = (bf16*)(ws + 144 * MB);          // [1024,512]
  bf16*  w2t   = (bf16*)(ws + 145 * MB);          // [512,1024]
  bf16*  wqt   = (bf16*)(ws + 146 * MB);          // [512,512]
  bf16*  wkt   = (bf16*)(ws + 146 * MB + 512 * 1024);
  bf16*  wvt   = (bf16*)(ws + 147 * MB);
  bf16*  gwt   = (bf16*)(ws + 147 * MB + 512 * 1024);
  bf16*  f1wt  = (bf16*)(ws + 148 * MB);          // [256,512]
  bf16*  f2wt  = (bf16*)(ws + 148 * MB + 256 * 1024);  // [128,256]

  auto wtgrid = [](int K, int Npad) { return (Npad * K + 255) / 256; };

  // 0) weight transpose+convert (tiny; L2 resident afterwards) + x -> bf16
  wt_cvt<<<wtgrid(512, 1024), 256, 0, stream>>>(fe_w1, w1t, 512, 1024, 1024);
  wt_cvt<<<wtgrid(1024, 512), 256, 0, stream>>>(fe_w2, w2t, 1024, 512, 512);
  wt_cvt<<<wtgrid(512, 512), 256, 0, stream>>>(wq, wqt, 512, 512, 512);
  wt_cvt<<<wtgrid(512, 512), 256, 0, stream>>>(wk, wkt, 512, 512, 512);
  wt_cvt<<<wtgrid(512, 512), 256, 0, stream>>>(wv, wvt, 512, 512, 512);
  wt_cvt<<<wtgrid(512, 512), 256, 0, stream>>>(gate_w, gwt, 512, 512, 512);
  wt_cvt<<<wtgrid(512, 256), 256, 0, stream>>>(fc1_w, f1wt, 512, 256, 256);
  wt_cvt<<<wtgrid(256, 128), 256, 0, stream>>>(fc2_w, f2wt, 256, 88, 128);
  f2bf<<<(M * 512 + 255) / 256, 256, 0, stream>>>(x, xb, (long)M * 512);

  const int rowsBlocks = M / 128;
  // 1) FE layer 1: h1b = gelu(x @ fe_w1 + b1)             [bf16 out]
  gemm_bf16t<512, EP_BIAS_GELU, bf16><<<dim3(1024 / 64, rowsBlocks), 256, 0, stream>>>(
      xb, w1t, fe_b1, nullptr, nullptr, h1b, M, 1024);
  // 2) FE layer 2 + residual: xln = h1 @ fe_w2 + b2 + x   [f32 out]
  gemm_bf16t<1024, EP_BIAS_RES, float><<<dim3(512 / 64, rowsBlocks), 256, 0, stream>>>(
      h1b, w2t, fe_b2, x, nullptr, xln, M, 512);
  // 3) LayerNorm -> bf16
  ln_rows<false><<<M, 256, 0, stream>>>(xln, xlnb, fe_g, fe_be, 512);
  // 4) QKV projections (V scattered into transposed layout)
  gemm_bf16t<512, EP_NONE, bf16><<<dim3(512 / 64, rowsBlocks), 256, 0, stream>>>(
      xlnb, wqt, nullptr, nullptr, nullptr, qb, M, 512);
  gemm_bf16t<512, EP_NONE, bf16><<<dim3(512 / 64, rowsBlocks), 256, 0, stream>>>(
      xlnb, wkt, nullptr, nullptr, nullptr, kb, M, 512);
  gemm_bf16t<512, EP_STORE_VT, bf16><<<dim3(512 / 64, rowsBlocks), 256, 0, stream>>>(
      xlnb, wvt, nullptr, nullptr, nullptr, vtb, M, 512);
  // 5) Flash attention (causal + decay-underflow mask) -> bf16
  attn_flash_wmma<<<dim3(16 * 4, 1024 / 64), 128, 0, stream>>>(qb, kb, vtb, attnb);
  // 6) Gate: gated = attn * sigmoid(attn @ gate_w + gate_b)
  gemm_bf16t<512, EP_GATE, bf16><<<dim3(512 / 64, rowsBlocks), 256, 0, stream>>>(
      attnb, gwt, gate_b, nullptr, attnb, gatedb, M, 512);
  // 7) fc1 (+bias) -> f32, then LN + GELU -> bf16
  gemm_bf16t<512, EP_BIAS, float><<<dim3(256 / 64, rowsBlocks), 256, 0, stream>>>(
      gatedb, f1wt, fc1_b, nullptr, nullptr, f1, M, 256);
  ln_rows<true><<<M, 256, 0, stream>>>(f1, f1b, ln_g, ln_b, 256);
  // 8) fc2 + sigmoid -> d_out (N=88 inside 128-padded tiles)
  gemm_bf16t<256, EP_BIAS_SIG, float><<<dim3(128 / 64, rowsBlocks), 256, 0, stream>>>(
      f1b, f2wt, fc2_b, nullptr, nullptr, out, M, 88);
}
